// TTTMamba2Refiner_70738111365401
// MI455X (gfx1250) — compile-verified, matrix operand order checked by reading, not run on hardware
//
#include <hip/hip_runtime.h>
#include <hip/hip_bf16.h>
#include <cstdint>
#include <math.h>

// ---------------- types ----------------
typedef _Float16 h16;
typedef h16 v16h __attribute__((ext_vector_type(16)));
typedef h16 v8h  __attribute__((ext_vector_type(8)));
typedef float v8f __attribute__((ext_vector_type(8)));

#define T_TOK 32768
#define HWSP  4096
#define NBATCH 8

__device__ __forceinline__ float geluf(float x){ return 0.5f*x*(1.0f+erff(x*0.7071067811865476f)); }
__device__ __forceinline__ float siluf(float x){ return x/(1.0f+__expf(-x)); }
__device__ __forceinline__ float softplusf(float x){ return (x>20.f)?x:log1pf(__expf(x)); }

// ---------------- WMMA helpers ----------------
// 16-bit A/B fragment per CDNA5 ISA layout:
// lane L<16 : row M=L,   K in {kb..kb+7} U {kb+16..kb+23} with kb=0
// lane L>=16: row M=L-16, same but kb=8  -> two contiguous 8-half (16B) loads.
__device__ __forceinline__ v16h load_frag(const h16* base, int ld, int lane){
  int row = lane & 15;
  int kb  = (lane >> 4) << 3;
  const h16* p = base + (long)row*ld + kb;
  v8h lo = *(const v8h*)p;
  v8h hi = *(const v8h*)(p + 16);
  v16h f;
#pragma unroll
  for (int i=0;i<8;++i){ f[i]=lo[i]; f[i+8]=hi[i]; }
  return f;
}
__device__ __forceinline__ v8f wmma16(v16h a, v16h b, v8f c){
  return __builtin_amdgcn_wmma_f32_16x16x32_f16(false, a, false, b, (short)0, c, false, false);
}
__device__ __forceinline__ v8f vzero8(){ v8f z = {0.f,0.f,0.f,0.f,0.f,0.f,0.f,0.f}; return z; }

// ---------------- generic batched GEMM: C[M,N] = A[M,K] * W[N,K]^T ----------------
// block = 128 threads = 4 waves stacked in M (block tile 128x64), wave tile 32x64.
// M%128==0, N%64==0, K%32==0.
__global__ void gemm_f16_kernel(const h16* __restrict__ A, const h16* __restrict__ W,
                                float* __restrict__ C, int K,
                                int lda, int ldb, int ldc,
                                long sA, long sB, long sC)
{
  int lane = threadIdx.x & 31;
  int wave = threadIdx.x >> 5;
  int z = blockIdx.z;
  const h16* Ab = A + z*sA + ((long)blockIdx.x*128 + wave*32)*lda;
  const h16* Wb = W + z*sB + (long)blockIdx.y*64*ldb;
  v8f acc[2][4];
#pragma unroll
  for (int i=0;i<2;++i)
#pragma unroll
    for (int j=0;j<4;++j) acc[i][j] = vzero8();

  for (int k0 = 0; k0 < K; k0 += 32){
    __builtin_prefetch(Ab + k0 + 64, 0, 1);           // global_prefetch_b8
    __builtin_prefetch(Wb + k0 + 64, 0, 1);
    v16h a0 = load_frag(Ab + k0,            lda, lane);
    v16h a1 = load_frag(Ab + 16*lda + k0,   lda, lane);
    v16h b0 = load_frag(Wb + k0,            ldb, lane);
    v16h b1 = load_frag(Wb + 16*ldb + k0,   ldb, lane);
    v16h b2 = load_frag(Wb + 32*ldb + k0,   ldb, lane);
    v16h b3 = load_frag(Wb + 48*ldb + k0,   ldb, lane);
    acc[0][0]=wmma16(a0,b0,acc[0][0]); acc[0][1]=wmma16(a0,b1,acc[0][1]);
    acc[0][2]=wmma16(a0,b2,acc[0][2]); acc[0][3]=wmma16(a0,b3,acc[0][3]);
    acc[1][0]=wmma16(a1,b0,acc[1][0]); acc[1][1]=wmma16(a1,b1,acc[1][1]);
    acc[1][2]=wmma16(a1,b2,acc[1][2]); acc[1][3]=wmma16(a1,b3,acc[1][3]);
  }
  float* Cb = C + z*sC + ((long)blockIdx.x*128 + wave*32)*ldc + (long)blockIdx.y*64;
  int r0 = (lane>>4)<<3;
  int col = lane & 15;
#pragma unroll
  for (int i=0;i<2;++i)
#pragma unroll
    for (int j=0;j<4;++j)
#pragma unroll
      for (int q=0;q<8;++q)
        Cb[(long)(i*16 + r0 + q)*ldc + j*16 + col] = acc[i][j][q];
}

// ---------------- packing ----------------
// dst[Np][Kp] f16, zero-padded. tr==0: src[N][K]; tr==1: src[K][N].
__global__ void pack_w_kernel(const float* __restrict__ src, h16* __restrict__ dst,
                              int N, int K, int Np, int Kp, int tr)
{
  long i = (long)blockIdx.x*blockDim.x + threadIdx.x;
  if (i >= (long)Np*Kp) return;
  int n = (int)(i / Kp), k = (int)(i % Kp);
  float v = 0.f;
  if (n < N && k < K) v = tr ? src[(long)k*N + n] : src[(long)n*K + k];
  dst[i] = (h16)v;
}

// NCHW fp32 -> token-major f16 [T][768]
__global__ void pack_feat_kernel(const float* __restrict__ src, h16* __restrict__ dst)
{
  long i = (long)blockIdx.x*blockDim.x + threadIdx.x;
  if (i >= (long)T_TOK*768) return;
  long t = i / 768; int c = (int)(i % 768);
  int b = (int)(t / HWSP), sp = (int)(t % HWSP);
  dst[i] = (h16)src[((long)(b*768 + c))*HWSP + sp];
}

// ---------------- batch GroupNorm ----------------
__global__ void reduce_mv_kernel(const float* __restrict__ X, int ldx, int c0, int C,
                                 float* __restrict__ mv)
{
  __shared__ float s1[256], s2[256];
  int b = blockIdx.x, tid = threadIdx.x;
  long n = (long)HWSP * C;
  float sum = 0.f, ss = 0.f;
  for (long i = tid; i < n; i += 256){
    long t = (long)b*HWSP + i / C;
    int  c = c0 + (int)(i % C);
    float v = X[t*ldx + c];
    sum += v; ss += v*v;
  }
  s1[tid]=sum; s2[tid]=ss; __syncthreads();
  for (int st=128; st>0; st>>=1){
    if (tid<st){ s1[tid]+=s1[tid+st]; s2[tid]+=s2[tid+st]; }
    __syncthreads();
  }
  if (tid==0){
    float m = s1[0]/(float)n;
    float v = s2[0]/(float)n - m*m;
    mv[b*2] = m; mv[b*2+1] = rsqrtf(v + 1e-5f);
  }
}

__global__ void gn_act_kernel(const float* __restrict__ X, int ldx, int c0, int C,
                              const float* __restrict__ mv,
                              const float* __restrict__ g, const float* __restrict__ bta,
                              const float* __restrict__ conf,
                              float* __restrict__ outF, int ldoF,
                              h16* __restrict__ outH, int ldoH, int o0H,
                              int do_gelu)
{
  long i = (long)blockIdx.x*blockDim.x + threadIdx.x;
  if (i >= (long)T_TOK*C) return;
  long t = i / C; int c = (int)(i % C);
  int b = (int)(t / HWSP);
  float v = (X[t*ldx + c0 + c] - mv[b*2]) * mv[b*2+1] * g[c] + bta[c];
  if (do_gelu) v = geluf(v);
  if (conf)    v *= (0.5f + 0.5f*conf[t]);
  if (outF) outF[t*ldoF + c] = v;
  if (outH) outH[t*ldoH + o0H + c] = (h16)v;
}

// ---------------- FiLM: denc -> MLP -> gamma/beta, apply to gelu(gn(film_proj)) ----------------
__global__ void film_kernel(const float* __restrict__ convOut,   // [T][384], film cols 0..191
                            const float* __restrict__ mv,
                            const float* __restrict__ gn_g, const float* __restrict__ gn_b,
                            const float* __restrict__ depth, const float* __restrict__ grads,
                            const float* __restrict__ w1, const float* __restrict__ b1,
                            const float* __restrict__ w2, const float* __restrict__ b2,
                            h16* __restrict__ fusIn)              // [T][576], write cols 192..383
{
  long t = (long)blockIdx.x*blockDim.x + threadIdx.x;
  if (t >= T_TOK) return;
  int b = (int)(t / HWSP), sp = (int)(t % HWSP);
  const float PI = 3.14159265358979323846f;
  float d  = depth[(long)b*HWSP + sp];
  float gx = grads[((long)(b*2+0))*HWSP + sp];
  float gy = grads[((long)(b*2+1))*HWSP + sp];
  float denc[15];
#pragma unroll
  for (int i=0;i<6;++i){
    float f = ldexpf(PI, i) * d;   // 2^i * pi * d
    denc[i]   = sinf(f);
    denc[6+i] = cosf(f);
  }
  denc[12]=d; denc[13]=gx; denc[14]=gy;
  float hbuf[64];
  for (int o=0;o<64;++o){
    float s = b1[o];
#pragma unroll
    for (int k=0;k<15;++k) s += w1[o*15+k]*denc[k];
    hbuf[o] = geluf(s);
  }
  float m = mv[b*2], r = mv[b*2+1];
  for (int o=0;o<192;++o){
    float ga = b2[o], be = b2[192+o];
    for (int k=0;k<64;++k){
      ga += w2[(long)o*64 + k]       * hbuf[k];
      be += w2[(long)(192+o)*64 + k] * hbuf[k];
    }
    ga = fminf(2.f, fmaxf(-2.f, ga));
    be = fminf(2.f, fmaxf(-2.f, be));
    float feat = geluf((convOut[t*384 + o] - m)*r*gn_g[o] + gn_b[o]);
    fusIn[t*576 + 192 + o] = (h16)(feat*(1.f+ga) + be);
  }
}

// ---------------- ctx conv 27->192 + conf ----------------
__global__ void ctx_conv_kernel(const float* __restrict__ probs, const float* __restrict__ w,
                                float* __restrict__ ctxRaw, float* __restrict__ conf)
{
  __shared__ float pr[32];
  long t = blockIdx.x;
  int b = (int)(t / HWSP), sp = (int)(t % HWSP);
  int tid = threadIdx.x;
  if (tid < 27) pr[tid] = probs[((long)(b*27 + tid))*HWSP + sp];
  __syncthreads();
  float s = 0.f;
#pragma unroll
  for (int k=0;k<27;++k) s += w[tid*27 + k]*pr[k];
  ctxRaw[t*192 + tid] = s;
  if (tid == 0){
    float mx = pr[0];
    for (int k=1;k<27;++k) mx = fmaxf(mx, pr[k]);
    conf[t] = mx;
  }
}

// ---------------- per-token LayerNorm (C=192) -> f16 ----------------
__global__ void ln_kernel(const float* __restrict__ xf, const float* __restrict__ g,
                          const float* __restrict__ b, h16* __restrict__ out)
{
  __shared__ float s1[256], s2[256];
  long t = blockIdx.x; int c = threadIdx.x;
  float x = xf[t*192 + c];
  s1[c]=x; s2[c]=x*x;
  if (c < 64){ s1[192+c]=0.f; s2[192+c]=0.f; }
  __syncthreads();
  for (int st=128; st>0; st>>=1){
    if (c<st){ s1[c]+=s1[c+st]; s2[c]+=s2[c+st]; }
    __syncthreads();
  }
  float m = s1[0]/192.f;
  float v = s2[0]/192.f - m*m;
  float r = rsqrtf(v + 1e-5f);
  out[t*192 + c] = (h16)((x-m)*r*g[c] + b[c]);
}

// ---------------- depthwise conv1d (D_CONV=4) + silu, + dt softplus ----------------
__global__ void conv1d_kernel(const float* __restrict__ zx, const float* __restrict__ cw,
                              const float* __restrict__ cb, const float* __restrict__ dtb,
                              h16* __restrict__ xBCh, float* __restrict__ dts)
{
  long t = blockIdx.x; int ch = threadIdx.x;
  int sp = (int)(t % HWSP);
  float acc = cb[ch];
#pragma unroll
  for (int j=0;j<4;++j){
    int pos = sp - 3 + j;
    if (pos >= 0) acc += zx[(t - 3 + j)*960 + 384 + ch] * cw[ch*4 + j];
  }
  xBCh[t*512 + ch] = (h16)siluf(acc);
  if (ch < 6){
    float raw = zx[t*960 + 896 + ch];
    dts[t*8 + ch] = softplusf(raw + dtb[ch]);
  }
}

// ---------------- per-chunk inclusive cumsum of dA = dt*A ----------------
__global__ void cumsum_kernel(const float* __restrict__ dts, const float* __restrict__ A_log,
                              float* __restrict__ dAcs)
{
  __shared__ float s[256];
  int z = blockIdx.x, h = blockIdx.y, c = threadIdx.x;
  long t = (long)z*256 + c;
  float A = -__expf(A_log[h]);
  s[c] = dts[t*8 + h] * A;
  __syncthreads();
  for (int off=1; off<256; off<<=1){
    float add = (c >= off) ? s[c-off] : 0.f;
    __syncthreads();
    s[c] += add;
    __syncthreads();
  }
  dAcs[t*8 + h] = s[c];
}

// ---------------- SSD Ydiag: Y[c,p] = sum_{s<=c} CB[c,s]*exp(dAcs[c]-dAcs[s])*dt[s]*x[s,p] ----------------
__global__ void ssd_ydiag_kernel(const float* __restrict__ CB, const float* __restrict__ dAcs,
                                 const float* __restrict__ dts, const h16* __restrict__ xBCh,
                                 float* __restrict__ Y)
{
  __shared__ h16 G[256*32];
  __shared__ h16 X[64*32];
  int z = blockIdx.x, h = blockIdx.y;
  int tid = threadIdx.x, lane = tid & 31, wave = tid >> 5;
  long t0 = (long)z*256;
  const float* cb = CB + (long)z*256*256;
  v8f acc[2][4];
#pragma unroll
  for (int i=0;i<2;++i)
#pragma unroll
    for (int j=0;j<4;++j) acc[i][j] = vzero8();

  float dac = dAcs[(t0 + tid)*8 + h];    // c = tid
  for (int j=0;j<8;++j){
    // build G tile [256 c][32 s] (f16) and X tile [64 p][32 s]
#pragma unroll
    for (int ss=0; ss<32; ++ss){
      int s = j*32 + ss;
      float g = 0.f;
      if (s <= tid){
        float das = dAcs[(t0 + s)*8 + h];
        g = cb[(long)tid*256 + s] * __expf(dac - das) * dts[(t0 + s)*8 + h];
      }
      G[tid*32 + ss] = (h16)g;
    }
#pragma unroll
    for (int q=0;q<8;++q){
      int e = tid*8 + q;
      int p = e >> 5, ss = e & 31;
      X[p*32 + ss] = xBCh[(t0 + j*32 + ss)*512 + h*64 + p];
    }
    __syncthreads();
    if (j <= wave){                       // causal tile skip
      v16h a0 = load_frag(&G[(wave*32)*32],    32, lane);
      v16h a1 = load_frag(&G[(wave*32+16)*32], 32, lane);
#pragma unroll
      for (int jn=0; jn<4; ++jn){
        v16h bfr = load_frag(&X[(jn*16)*32], 32, lane);
        acc[0][jn] = wmma16(a0, bfr, acc[0][jn]);
        acc[1][jn] = wmma16(a1, bfr, acc[1][jn]);
      }
    }
    __syncthreads();
  }
  int r0 = (lane>>4)<<3, col = lane & 15;
#pragma unroll
  for (int i=0;i<2;++i)
#pragma unroll
    for (int jn=0;jn<4;++jn)
#pragma unroll
      for (int q=0;q<8;++q)
        Y[(t0 + wave*32 + i*16 + r0 + q)*384 + h*64 + jn*16 + col] = acc[i][jn][q];
}

// ---------------- SSD states: S[n,p] = sum_c B[c,n]*exp(dAcs_L - dAcs[c])*dt[c]*x[c,p] ----------------
__global__ void ssd_states_kernel(const h16* __restrict__ xBCh, const float* __restrict__ dAcs,
                                  const float* __restrict__ dts, float* __restrict__ states)
{
  __shared__ h16 Ash[64*32];
  __shared__ h16 X[64*32];
  int z = blockIdx.x, h = blockIdx.y;
  int tid = threadIdx.x, lane = tid & 31, wave = tid >> 5;   // 64 threads, 2 waves
  long t0 = (long)z*256;
  float daL = dAcs[(t0 + 255)*8 + h];
  v8f acc[2][4];
#pragma unroll
  for (int i=0;i<2;++i)
#pragma unroll
    for (int j=0;j<4;++j) acc[i][j] = vzero8();

  for (int j=0;j<8;++j){
    // thread tid builds row n=tid / p=tid (64 rows x 32 cols each)
#pragma unroll
    for (int q=0;q<32;++q){
      int c = j*32 + q;
      float wgt = __expf(daL - dAcs[(t0+c)*8+h]) * dts[(t0+c)*8+h];
      Ash[tid*32 + q] = (h16)((float)xBCh[(t0+c)*512 + 384 + tid] * wgt);
      X[tid*32 + q]   = xBCh[(t0+c)*512 + h*64 + tid];
    }
    __syncthreads();
    v16h a0 = load_frag(&Ash[(wave*32)*32],    32, lane);
    v16h a1 = load_frag(&Ash[(wave*32+16)*32], 32, lane);
#pragma unroll
    for (int jn=0; jn<4; ++jn){
      v16h bfr = load_frag(&X[(jn*16)*32], 32, lane);
      acc[0][jn] = wmma16(a0, bfr, acc[0][jn]);
      acc[1][jn] = wmma16(a1, bfr, acc[1][jn]);
    }
    __syncthreads();
  }
  float* sb = states + ((long)z*6 + h)*4096;
  int r0 = (lane>>4)<<3, col = lane & 15;
#pragma unroll
  for (int i=0;i<2;++i)
#pragma unroll
    for (int jn=0;jn<4;++jn)
#pragma unroll
      for (int q=0;q<8;++q)
        sb[(long)(wave*32 + i*16 + r0 + q)*64 + jn*16 + col] = acc[i][jn][q];
}

// ---------------- inter-chunk scan: prev[z] = carry; carry = carry*cd[z] + states[z] ----------------
__global__ void ssd_scan_kernel(const float* __restrict__ states, const float* __restrict__ dAcs,
                                float* __restrict__ prev)
{
  int b = blockIdx.x / 6, h = blockIdx.x % 6;
  int tid = threadIdx.x;
  float carry[16];
#pragma unroll
  for (int q=0;q<16;++q) carry[q] = 0.f;
  for (int zl=0; zl<16; ++zl){
    int zz = b*16 + zl;
    float cd = __expf(dAcs[((long)zz*256 + 255)*8 + h]);
    long base = ((long)zz*6 + h)*4096;
#pragma unroll
    for (int q=0;q<16;++q){
      long e = base + tid*16 + q;
      prev[e]  = carry[q];
      carry[q] = carry[q]*cd + states[e];
    }
  }
}

// ---------------- SSD Yoff: Y[c,p] += sum_n C[c,n]*exp(dAcs[c]) * prev[n,p] ----------------
__global__ void ssd_yoff_kernel(const h16* __restrict__ xBCh, const float* __restrict__ dAcs,
                                const float* __restrict__ prev, float* __restrict__ Y)
{
  __shared__ h16 Ash[256*32];
  __shared__ h16 Xp[64*64];
  int z = blockIdx.x, h = blockIdx.y;
  int tid = threadIdx.x, lane = tid & 31, wave = tid >> 5;
  long t0 = (long)z*256;
  const float* pv = prev + ((long)z*6 + h)*4096;
  // transpose prev [n][p] -> Xp[p][n]
#pragma unroll
  for (int q=0;q<16;++q){
    int e = tid*16 + q;
    int p = e >> 6, n = e & 63;
    Xp[p*64 + n] = (h16)pv[(long)n*64 + p];
  }
  float sc = __expf(dAcs[(t0 + tid)*8 + h]);   // c = tid
  v8f acc[2][4];
#pragma unroll
  for (int i=0;i<2;++i)
#pragma unroll
    for (int j=0;j<4;++j) acc[i][j] = vzero8();
  __syncthreads();

  for (int j=0;j<2;++j){
#pragma unroll
    for (int q=0;q<32;++q)
      Ash[tid*32 + q] = (h16)((float)xBCh[(t0 + tid)*512 + 448 + j*32 + q] * sc);
    __syncthreads();
    v16h a0 = load_frag(&Ash[(wave*32)*32],    32, lane);
    v16h a1 = load_frag(&Ash[(wave*32+16)*32], 32, lane);
#pragma unroll
    for (int jn=0; jn<4; ++jn){
      v16h bfr = load_frag(&Xp[(jn*16)*64 + j*32], 64, lane);
      acc[0][jn] = wmma16(a0, bfr, acc[0][jn]);
      acc[1][jn] = wmma16(a1, bfr, acc[1][jn]);
    }
    __syncthreads();
  }
  int r0 = (lane>>4)<<3, col = lane & 15;
#pragma unroll
  for (int i=0;i<2;++i)
#pragma unroll
    for (int jn=0;jn<4;++jn)
#pragma unroll
      for (int q=0;q<8;++q)
        Y[(t0 + wave*32 + i*16 + r0 + q)*384 + h*64 + jn*16 + col] += acc[i][jn][q];
}

// ---------------- combine: (Y + D*x)*silu(z), RMSNorm*rms_w -> f16 ----------------
__global__ void combine_y_kernel(const float* __restrict__ Y, const h16* __restrict__ xBCh,
                                 const float* __restrict__ zx, const float* __restrict__ D,
                                 const float* __restrict__ rmsw, h16* __restrict__ yh)
{
  __shared__ float s[512];
  long t = blockIdx.x; int hp = threadIdx.x;
  int h = hp >> 6;
  float y = Y[t*384 + hp] + D[h]*(float)xBCh[t*512 + hp];
  y *= siluf(zx[t*960 + hp]);
  s[hp] = y*y;
  if (hp < 128) s[384 + hp] = 0.f;
  __syncthreads();
  for (int st=256; st>0; st>>=1){
    if (hp < st) s[hp] += s[hp+st];
    __syncthreads();
  }
  float r = rsqrtf(s[0]/384.f + 1e-5f);
  yh[t*384 + hp] = (h16)(y * r * rmsw[hp]);
}

// ---------------- misc elementwise ----------------
__global__ void add_kernel(float* __restrict__ xf, const float* __restrict__ d, int ldd,
                           const float* __restrict__ bias)
{
  long i = (long)blockIdx.x*blockDim.x + threadIdx.x;
  if (i >= (long)T_TOK*192) return;
  long t = i / 192; int c = (int)(i % 192);
  float v = d[t*ldd + c];
  if (bias) v += bias[c];
  xf[i] += v;
}

__global__ void bias_gelu_kernel(const float* __restrict__ in, const float* __restrict__ b,
                                 h16* __restrict__ out)
{
  long i = (long)blockIdx.x*blockDim.x + threadIdx.x;
  if (i >= (long)T_TOK*384) return;
  int c = (int)(i % 384);
  out[i] = (h16)geluf(in[i] + b[c]);
}

__global__ void head_write_kernel(const float* __restrict__ hOut, const float* __restrict__ hb,
                                  float* __restrict__ out)
{
  long i = (long)blockIdx.x*blockDim.x + threadIdx.x;
  if (i >= (long)T_TOK*27) return;
  long t = i / 27; int o = (int)(i % 27);
  int b = (int)(t / HWSP), sp = (int)(t % HWSP);
  out[((long)(b*27 + o))*HWSP + sp] = hOut[t*64 + o] + hb[o];
}

// ================= host =================
extern "C" void kernel_launch(void* const* d_in, const int* in_sizes, int n_in,
                              void* d_out, int out_size, void* d_ws, size_t ws_size,
                              hipStream_t stream)
{
  (void)in_sizes; (void)n_in; (void)out_size; (void)ws_size;
  auto P = [&](int i){ return (const float*)d_in[i]; };
  const float* feats  = P(0);
  const float* depth  = P(1);
  const float* grads  = P(2);
  const float* probs  = P(3);
  // params tree-leaves: blocks(3x16), ctx(3), feat(3), film(7), fusion(3), head(4)
  const int BLK0 = 4, BS = 16;
  // within-block offsets: A_log,D,conv_b,conv_w,dt_bias,ffn_b1,ffn_b2,ffn_w1,ffn_w2,
  //                       in_proj,ln1_b,ln1_g,ln2_b,ln2_g,out_proj,rms_w
  const float* ctx_gnb = P(52); const float* ctx_gng = P(53); const float* ctx_w = P(54);
  const float* fe_gnb  = P(55); const float* fe_gng  = P(56); const float* fe_w  = P(57);
  const float* fi_b1   = P(58); const float* fi_b2   = P(59);
  const float* fi_w1   = P(60); const float* fi_w2   = P(61);
  const float* fi_gnb  = P(62); const float* fi_gng  = P(63); const float* fi_w   = P(64);
  const float* fu_gnb  = P(65); const float* fu_gng  = P(66); const float* fu_w   = P(67);
  const float* hd_b    = P(68); const float* hd_gnb  = P(69); const float* hd_gng = P(70);
  const float* hd_w    = P(71);

  uint8_t* wsb = (uint8_t*)d_ws; size_t off = 0;
  auto alloc = [&](size_t bytes)->void*{ void* p = wsb + off; off = (off + bytes + 255) & ~(size_t)255; return p; };
  const long T = T_TOK;
  h16*   featTok = (h16*)  alloc(T*768*2);
  h16*   Wcomb   = (h16*)  alloc(384L*768*2);
  float* featOut = (float*)alloc(T*384*4);
  h16*   fusIn   = (h16*)  alloc(T*576*2);
  h16*   Wfus    = (h16*)  alloc(192L*576*2);
  float* fusOut  = (float*)alloc(T*192*4);
  float* ctxRaw  = (float*)alloc(T*192*4);
  float* conf    = (float*)alloc(T*4);
  float* mv      = (float*)alloc(5*16*4);
  float* xf      = (float*)alloc(T*192*4);
  h16*   lnBuf   = (h16*)  alloc(T*192*2);
  h16*   Winp    = (h16*)  alloc(960L*192*2);
  h16*   Woutp   = (h16*)  alloc(192L*384*2);
  h16*   Wf1     = (h16*)  alloc(384L*192*2);
  h16*   Wf2     = (h16*)  alloc(192L*384*2);
  h16*   Whead   = (h16*)  alloc(64L*192*2);
  float* zx      = (float*)alloc(T*960*4);
  h16*   xBCh    = (h16*)  alloc(T*512*2);
  float* dts     = (float*)alloc(T*8*4);
  float* dAcs    = (float*)alloc(T*8*4);
  float* CBb     = (float*)alloc(128L*256*256*4);
  float* Ybuf    = (float*)alloc(T*384*4);
  float* states  = (float*)alloc(128L*6*4096*4);
  float* prev    = (float*)alloc(128L*6*4096*4);
  h16*   yh      = (h16*)  alloc(T*384*2);
  float* tmpOut  = (float*)alloc(T*384*4);
  h16*   gBuf    = (h16*)  alloc(T*384*2);
  float* headOut = (float*)alloc(T*64*4);

  auto packw = [&](const float* s, h16* d, int N, int K, int Np, int Kp, int tr){
    long n = (long)Np*Kp;
    pack_w_kernel<<<(unsigned)((n+255)/256), 256, 0, stream>>>(s, d, N, K, Np, Kp, tr);
  };
  auto gemm = [&](const h16* A, const h16* W, float* C, int M, int N, int K,
                  int lda, int ldb, int ldc, int batch, long sA, long sB, long sC){
    dim3 g(M/128, N/64, batch);
    gemm_f16_kernel<<<g, 128, 0, stream>>>(A, W, C, K, lda, ldb, ldc, sA, sB, sC);
  };

  // ---- features path ----
  pack_feat_kernel<<<(unsigned)((T*768+255)/256), 256, 0, stream>>>(feats, featTok);
  packw(fi_w, Wcomb,         192, 768, 192, 768, 0);   // film proj -> rows 0..191
  packw(fe_w, Wcomb+192*768, 192, 768, 192, 768, 0);   // feat sem  -> rows 192..383
  gemm(featTok, Wcomb, featOut, T_TOK, 384, 768, 768, 768, 384, 1, 0, 0, 0);

  reduce_mv_kernel<<<NBATCH, 256, 0, stream>>>(featOut, 384,   0, 192, mv + 0*16);
  reduce_mv_kernel<<<NBATCH, 256, 0, stream>>>(featOut, 384, 192, 192, mv + 1*16);
  // feat_sem -> fusIn[:,0:192]
  gn_act_kernel<<<(unsigned)((T*192+255)/256), 256, 0, stream>>>(
      featOut, 384, 192, 192, mv + 1*16, fe_gng, fe_gnb, nullptr,
      nullptr, 0, fusIn, 576, 0, 1);
  // FiLM -> fusIn[:,192:384]
  film_kernel<<<(unsigned)((T+127)/128), 128, 0, stream>>>(
      featOut, mv + 0*16, fi_gng, fi_gnb, depth, grads, fi_w1, fi_b1, fi_w2, fi_b2, fusIn);

  // ---- ctx path -> fusIn[:,384:576] ----
  ctx_conv_kernel<<<(unsigned)T_TOK, 192, 0, stream>>>(probs, ctx_w, ctxRaw, conf);
  reduce_mv_kernel<<<NBATCH, 256, 0, stream>>>(ctxRaw, 192, 0, 192, mv + 2*16);
  gn_act_kernel<<<(unsigned)((T*192+255)/256), 256, 0, stream>>>(
      ctxRaw, 192, 0, 192, mv + 2*16, ctx_gng, ctx_gnb, conf,
      nullptr, 0, fusIn, 576, 384, 1);

  // ---- fusion -> xf ----
  packw(fu_w, Wfus, 192, 576, 192, 576, 0);
  gemm(fusIn, Wfus, fusOut, T_TOK, 192, 576, 576, 576, 192, 1, 0, 0, 0);
  reduce_mv_kernel<<<NBATCH, 256, 0, stream>>>(fusOut, 192, 0, 192, mv + 3*16);
  gn_act_kernel<<<(unsigned)((T*192+255)/256), 256, 0, stream>>>(
      fusOut, 192, 0, 192, mv + 3*16, fu_gng, fu_gnb, nullptr,
      xf, 192, nullptr, 0, 0, 1);

  // ---- mamba blocks ----
  for (int bi = 0; bi < 3; ++bi){
    int base = BLK0 + bi*BS;
    const float* A_log = P(base+0); const float* Dp   = P(base+1);
    const float* cvb   = P(base+2); const float* cvw  = P(base+3);
    const float* dtb   = P(base+4); const float* fb1  = P(base+5);
    const float* fb2   = P(base+6); const float* fw1  = P(base+7);
    const float* fw2   = P(base+8); const float* inp  = P(base+9);
    const float* l1b   = P(base+10); const float* l1g = P(base+11);
    const float* l2b   = P(base+12); const float* l2g = P(base+13);
    const float* outp  = P(base+14); const float* rmsw= P(base+15);

    packw(inp,  Winp,  902, 192, 960, 192, 1);   // (192,902) K-major -> [960][192]
    packw(outp, Woutp, 192, 384, 192, 384, 1);
    packw(fw1,  Wf1,   384, 192, 384, 192, 1);
    packw(fw2,  Wf2,   192, 384, 192, 384, 1);

    ln_kernel<<<(unsigned)T_TOK, 192, 0, stream>>>(xf, l1g, l1b, lnBuf);
    gemm(lnBuf, Winp, zx, T_TOK, 960, 192, 192, 192, 960, 1, 0, 0, 0);
    conv1d_kernel<<<(unsigned)T_TOK, 512, 0, stream>>>(zx, cvw, cvb, dtb, xBCh, dts);
    { dim3 g(128, 6); cumsum_kernel<<<g, 256, 0, stream>>>(dts, A_log, dAcs); }

    // CB[c,s] per chunk: A=C-slice(col 448), W=B-slice(col 384), both ld=512
    gemm(xBCh + 448, xBCh + 384, CBb, 256, 256, 64, 512, 512, 256,
         128, 256L*512, 256L*512, 256L*256);

    { dim3 g(128, 6);
      ssd_ydiag_kernel <<<g, 256, 0, stream>>>(CBb, dAcs, dts, xBCh, Ybuf);
      ssd_states_kernel<<<g,  64, 0, stream>>>(xBCh, dAcs, dts, states);
    }
    ssd_scan_kernel<<<NBATCH*6, 256, 0, stream>>>(states, dAcs, prev);
    { dim3 g(128, 6); ssd_yoff_kernel<<<g, 256, 0, stream>>>(xBCh, dAcs, prev, Ybuf); }

    combine_y_kernel<<<(unsigned)T_TOK, 384, 0, stream>>>(Ybuf, xBCh, zx, Dp, rmsw, yh);
    gemm(yh, Woutp, tmpOut, T_TOK, 192, 384, 384, 384, 192, 1, 0, 0, 0);
    add_kernel<<<(unsigned)((T*192+255)/256), 256, 0, stream>>>(xf, tmpOut, 192, nullptr);

    ln_kernel<<<(unsigned)T_TOK, 192, 0, stream>>>(xf, l2g, l2b, lnBuf);
    gemm(lnBuf, Wf1, tmpOut, T_TOK, 384, 192, 192, 192, 384, 1, 0, 0, 0);
    bias_gelu_kernel<<<(unsigned)((T*384+255)/256), 256, 0, stream>>>(tmpOut, fb1, gBuf);
    gemm(gBuf, Wf2, tmpOut, T_TOK, 192, 384, 384, 384, 192, 1, 0, 0, 0);
    add_kernel<<<(unsigned)((T*192+255)/256), 256, 0, stream>>>(xf, tmpOut, 192, fb2);
  }

  // ---- head: gn -> conv1x1 27 ----
  reduce_mv_kernel<<<NBATCH, 256, 0, stream>>>(xf, 192, 0, 192, mv + 4*16);
  gn_act_kernel<<<(unsigned)((T*192+255)/256), 256, 0, stream>>>(
      xf, 192, 0, 192, mv + 4*16, hd_gng, hd_gnb, nullptr,
      nullptr, 0, lnBuf, 192, 0, 0);
  packw(hd_w, Whead, 27, 192, 64, 192, 0);
  gemm(lnBuf, Whead, headOut, T_TOK, 64, 192, 192, 192, 64, 1, 0, 0, 0);
  head_write_kernel<<<(unsigned)((T*27+255)/256), 256, 0, stream>>>(headOut, hd_b, (float*)d_out);
}
